// MultiHeadCrossAttention_47519518163418
// MI455X (gfx1250) — compile-verified, matrix operand order checked by reading, not run on hardware
//
#include <hip/hip_runtime.h>
#include <hip/hip_bf16.h>

#define EPSN 1e-5f

typedef __attribute__((ext_vector_type(16))) __bf16 bf16x16;
typedef __attribute__((ext_vector_type(8)))  float  f32x8;
typedef __attribute__((ext_vector_type(8)))  unsigned short u16x8;
typedef __attribute__((ext_vector_type(4)))  int    i32x4;

#if __has_builtin(__builtin_amdgcn_global_load_async_to_lds_b128)
#define HAVE_ASYNC_LDS 1
#endif

#ifdef HAVE_ASYNC_LDS
typedef __attribute__((address_space(1))) i32x4* gv4p;
typedef __attribute__((address_space(3))) i32x4* lv4p;
__device__ __forceinline__ void async_b128(const unsigned short* g, unsigned short* l) {
  __builtin_amdgcn_global_load_async_to_lds_b128((gv4p)g, (lv4p)l, 0, 0);
}
__device__ __forceinline__ void wait_async0() {
#if __has_builtin(__builtin_amdgcn_s_wait_asynccnt)
  __builtin_amdgcn_s_wait_asynccnt(0);
#else
  asm volatile("s_wait_asynccnt 0" ::: "memory");
#endif
}
#endif

__device__ __forceinline__ unsigned short f2bf_bits(float f) {
  unsigned u = __builtin_bit_cast(unsigned, f);
  u += 0x7FFFu + ((u >> 16) & 1u);          // round-to-nearest-even
  return (unsigned short)(u >> 16);
}

__device__ __forceinline__ void load8(bf16x16& a, int base, const unsigned short* p) {
  u16x8 v = *(const u16x8*)p;               // 16B aligned load
#pragma unroll
  for (int i = 0; i < 8; ++i) a[base + i] = __builtin_bit_cast(__bf16, (unsigned short)v[i]);
}

__device__ __forceinline__ f32x8 wmma_bf16f32(bf16x16 a, bf16x16 b, f32x8 c) {
  return __builtin_amdgcn_wmma_f32_16x16x32_bf16(false, a, false, b, (short)0, c, false, false);
}

// ---------------------------------------------------------------------------
// Generic reflect-pad 3x3 conv as implicit GEMM (bf16 WMMA, f32 accum).
// Optional nearest-upsample-by-2 of the input and optional sin/cos positional
// encoding folded into the A-tile gather.  Cout fixed at 128.
// grid.x = B * OH * (OW/64), block = 128 (4 waves).
// ---------------------------------------------------------------------------
__global__ __launch_bounds__(128)
void conv3x3_wmma(const float* __restrict__ in, const unsigned short* __restrict__ wbf,
                  float* __restrict__ out, int Cin, int logH, int logW,
                  int OH, int OW, int stride, int ups, int addPE)
{
  const int K = Cin * 9;
  const int tilesPerRow = OW >> 6;
  const int tx  = blockIdx.x % tilesPerRow;
  const int oy  = (blockIdx.x / tilesPerRow) % OH;
  const int b   = blockIdx.x / (tilesPerRow * OH);
  const int ox0 = tx << 6;
  const int srcH = ups ? (logH >> 1) : logH;
  const int srcW = ups ? (logW >> 1) : logW;
  const int half = Cin >> 1;
  const float invHalf = 1.f / (float)half;  // computed once, kills per-elem fdiv

  __shared__ alignas(16) unsigned short Alds[64 * 32];
  __shared__ alignas(16) unsigned short Blds[128 * 32];

  const int tid    = threadIdx.x;
  const int wave   = tid >> 5;
  const int lane   = tid & 31;
  const int lane16 = lane & 15;
  const int khalf  = lane >> 4;
  const int kk     = tid & 31;              // invariant across staging loop
  const int mbase  = tid >> 5;              // 0..3

  f32x8 acc[8] = {};

  for (int kc = 0; kc < K; kc += 32) {
    __syncthreads();
    // ---- stage A tile: 64 (pixels) x 32 (k); k-decomposition hoisted per chunk
    {
      int k  = kc + kk;
      int ci = k / 9;
      int t  = k - ci * 9;
      int kh = t / 3;
      int kw = t - kh * 3;
      int iy = oy * stride + kh - 1;
      iy = iy < 0 ? -iy : (iy >= logH ? 2 * logH - 2 - iy : iy);
      const float* rowp = in + ((size_t)(b * Cin + ci) * srcH + (ups ? (iy >> 1) : iy)) * srcW;
      float rate = 0.f, rowAng = 0.f;
      bool useSin = false;
      if (addPE) {
        int cc = ci < half ? ci : ci - half;
        rate   = __expf(-9.210340371976184f * ((float)cc * invHalf)); // 10000^-d
        useSin = ci < half;
        rowAng = (float)(iy * logW) * rate;
      }
#pragma unroll
      for (int it = 0; it < 16; ++it) {
        int m  = mbase + 4 * it;
        int ix = (ox0 + m) * stride + kw - 1;
        ix = ix < 0 ? -ix : (ix >= logW ? 2 * logW - 2 - ix : ix);
        float val = rowp[ups ? (ix >> 1) : ix];
        if (addPE) {
          float ang = rowAng + (float)ix * rate;
          val += useSin ? __sinf(ang) : __cosf(ang);
        }
        Alds[m * 32 + kk] = f2bf_bits(val);
      }
    }
    // ---- stage B tile: 128 (cout) x 32 (k); bf16 weights are already [co][k]
    {
      const unsigned short* src = wbf + (size_t)tid * K + kc;
      unsigned short* dst = Blds + tid * 32;
#ifdef HAVE_ASYNC_LDS
#pragma unroll
      for (int i = 0; i < 4; ++i) async_b128(src + i * 8, dst + i * 8);
      wait_async0();
#else
#pragma unroll
      for (int i = 0; i < 4; ++i) ((u16x8*)dst)[i] = ((const u16x8*)src)[i];
#endif
    }
    __syncthreads();
    // ---- fragments + 8 WMMAs
    bf16x16 af;
    const unsigned short* ap = Alds + (wave * 16 + lane16) * 32 + khalf * 8;
    load8(af, 0, ap);
    load8(af, 8, ap + 16);
#pragma unroll
    for (int nc = 0; nc < 8; ++nc) {
      bf16x16 bfr;
      const unsigned short* bp = Blds + (nc * 16 + lane16) * 32 + khalf * 8;
      load8(bfr, 0, bp);
      load8(bfr, 8, bp + 16);
      acc[nc] = wmma_bf16f32(af, bfr, acc[nc]);
    }
  }
  // ---- store: C layout -> lane holds rows r+8*khalf, col lane16 of chunk nc
#pragma unroll
  for (int nc = 0; nc < 8; ++nc) {
    int n = nc * 16 + lane16;
#pragma unroll
    for (int r = 0; r < 8; ++r) {
      int m  = r + 8 * khalf;
      int ox = ox0 + wave * 16 + m;
      out[((size_t)(b * 128 + n) * OH + oy) * OW + ox] = acc[nc][r];
    }
  }
}

// ---------------------------------------------------------------------------
// Flash-style MHA: WG = (b, head, 64 query rows); never materializes NxN.
// q_bf: [B,N,C] bf16 (queries AND keys); vT_bf: [B,C,N] bf16; out: [B,C,N] f32
// ---------------------------------------------------------------------------
__global__ __launch_bounds__(128)
void flash_attn(const unsigned short* __restrict__ q_bf,
                const unsigned short* __restrict__ vT_bf,
                float* __restrict__ mha, int N, int C, int heads)
{
  const int d = C / heads;                  // 32
  const int nTiles = N >> 6;
  const int qt = blockIdx.x % nTiles;
  const int h  = (blockIdx.x / nTiles) % heads;
  const int b  = blockIdx.x / (nTiles * heads);

  const int tid    = threadIdx.x;
  const int wave   = tid >> 5;
  const int lane   = tid & 31;
  const int lane16 = lane & 15;
  const int khalf  = lane >> 4;
  const int qbase  = qt * 64 + wave * 16;

  __shared__ alignas(16) unsigned short Klds[32 * 32];   // [j][d]
  __shared__ alignas(16) unsigned short Vlds[32 * 32];   // [d][j]
  __shared__ alignas(16) unsigned short Plds[4][16 * 32];

  // Q fragment (A layout): row = qbase+lane16, K(=d) halves by khalf
  bf16x16 qf;
  const unsigned short* qrow = q_bf + ((size_t)(b * N + qbase + lane16)) * C + h * d;
  load8(qf, 0, qrow + khalf * 8);
  load8(qf, 8, qrow + 16 + khalf * 8);

  float mrun[8], lrun[8];
#pragma unroll
  for (int r = 0; r < 8; ++r) { mrun[r] = -1e30f; lrun[r] = 0.f; }
  f32x8 acc0 = {}, acc1 = {};
  const float scale = rsqrtf((float)d);

  const int srow = tid >> 2;                // 0..31
  const int sseg = (tid & 3) * 8;           // 0,8,16,24

  for (int j0 = 0; j0 < N; j0 += 32) {
    __syncthreads();
    {   // stage 32 keys + 32 value rows (this head) into LDS
      const unsigned short* ksrc = q_bf + ((size_t)(b * N + j0 + srow)) * C + h * d + sseg;
      const unsigned short* vsrc = vT_bf + ((size_t)(b * C + h * d + srow)) * N + j0 + sseg;
#ifdef HAVE_ASYNC_LDS
      async_b128(ksrc, Klds + srow * 32 + sseg);
      async_b128(vsrc, Vlds + srow * 32 + sseg);
      wait_async0();
#else
      *(u16x8*)(Klds + srow * 32 + sseg) = *(const u16x8*)ksrc;
      *(u16x8*)(Vlds + srow * 32 + sseg) = *(const u16x8*)vsrc;
#endif
    }
    __syncthreads();
    // ---- scores: two 16x16 tiles (cols j0..15, j0+16..31)
    f32x8 s0 = {}, s1 = {};
    bf16x16 kf;
    const unsigned short* kp0 = Klds + lane16 * 32 + khalf * 8;
    load8(kf, 0, kp0); load8(kf, 8, kp0 + 16);
    s0 = wmma_bf16f32(qf, kf, s0);
    const unsigned short* kp1 = Klds + (16 + lane16) * 32 + khalf * 8;
    load8(kf, 0, kp1); load8(kf, 8, kp1 + 16);
    s1 = wmma_bf16f32(qf, kf, s1);
    // ---- online softmax (rows r+8*khalf live across 16-lane groups)
    float p0[8], p1[8];
#pragma unroll
    for (int r = 0; r < 8; ++r) {
      float a = s0[r] * scale, bb = s1[r] * scale;
      float mx = fmaxf(a, bb);
      for (int msk = 1; msk <= 8; msk <<= 1) mx = fmaxf(mx, __shfl_xor(mx, msk, 32));
      float nm = fmaxf(mrun[r], mx);
      float e0 = __expf(a - nm), e1 = __expf(bb - nm);
      float rs = e0 + e1;
      for (int msk = 1; msk <= 8; msk <<= 1) rs += __shfl_xor(rs, msk, 32);
      float cr = __expf(mrun[r] - nm);
      lrun[r] = lrun[r] * cr + rs;
      mrun[r] = nm;
      p0[r] = e0; p1[r] = e1;
      acc0[r] *= cr; acc1[r] *= cr;
    }
    // ---- P (C layout) -> LDS -> reload in A layout
    unsigned short* pw = Plds[wave];
#pragma unroll
    for (int r = 0; r < 8; ++r) {
      int m = r + 8 * khalf;
      pw[m * 32 + lane16]      = f2bf_bits(p0[r]);
      pw[m * 32 + 16 + lane16] = f2bf_bits(p1[r]);
    }
    __syncthreads();
    bf16x16 pf;
    const unsigned short* pp = pw + lane16 * 32 + khalf * 8;
    load8(pf, 0, pp); load8(pf, 8, pp + 16);
    bf16x16 vf;
    const unsigned short* vp0 = Vlds + lane16 * 32 + khalf * 8;        // d cols 0..15
    load8(vf, 0, vp0); load8(vf, 8, vp0 + 16);
    acc0 = wmma_bf16f32(pf, vf, acc0);
    const unsigned short* vp1 = Vlds + (16 + lane16) * 32 + khalf * 8; // d cols 16..31
    load8(vf, 0, vp1); load8(vf, 8, vp1 + 16);
    acc1 = wmma_bf16f32(pf, vf, acc1);
  }
  // ---- normalize + store NCHW
#pragma unroll
  for (int r = 0; r < 8; ++r) {
    int m = r + 8 * khalf;
    float inv = 1.f / lrun[r];
    mha[((size_t)(b * C + h * d + lane16)) * N + qbase + m]      = acc0[r] * inv;
    mha[((size_t)(b * C + h * d + 16 + lane16)) * N + qbase + m] = acc1[r] * inv;
  }
}

// ---------------------------------------------------------------------------
// Small helper kernels
// ---------------------------------------------------------------------------
__global__ void cvt_bf16(const float* __restrict__ src, unsigned short* __restrict__ dst, int n) {
  int i = blockIdx.x * blockDim.x + threadIdx.x;
  if (i < n) dst[i] = f2bf_bits(src[i]);
}

__global__ void sum_stats(const float* __restrict__ x, float* __restrict__ sums,
                          float* __restrict__ ssqs, int HW) {
  int bc = blockIdx.x;
  const float* p = x + (size_t)bc * HW;
  float s = 0.f, ss = 0.f;
  for (int i = threadIdx.x; i < HW; i += blockDim.x) { float v = p[i]; s += v; ss += v * v; }
  __shared__ float shs[256], shq[256];
  shs[threadIdx.x] = s; shq[threadIdx.x] = ss;
  __syncthreads();
  for (int o = blockDim.x >> 1; o > 0; o >>= 1) {
    if ((int)threadIdx.x < o) { shs[threadIdx.x] += shs[threadIdx.x + o]; shq[threadIdx.x] += shq[threadIdx.x + o]; }
    __syncthreads();
  }
  if (threadIdx.x == 0) { sums[bc] = shs[0]; ssqs[bc] = shq[0]; }
}

__global__ void in_stats(const float* __restrict__ x, float* mean, float* invstd, int HW) {
  int bc = blockIdx.x;
  const float* p = x + (size_t)bc * HW;
  float s = 0.f, ss = 0.f;
  for (int i = threadIdx.x; i < HW; i += blockDim.x) { float v = p[i]; s += v; ss += v * v; }
  __shared__ float shs[256], shq[256];
  shs[threadIdx.x] = s; shq[threadIdx.x] = ss;
  __syncthreads();
  for (int o = blockDim.x >> 1; o > 0; o >>= 1) {
    if ((int)threadIdx.x < o) { shs[threadIdx.x] += shs[threadIdx.x + o]; shq[threadIdx.x] += shq[threadIdx.x + o]; }
    __syncthreads();
  }
  if (threadIdx.x == 0) {
    float m = shs[0] / HW;
    float v = shq[0] / HW - m * m;            // biased (InstanceNorm2d)
    mean[bc] = m; invstd[bc] = rsqrtf(v + EPSN);
  }
}

__global__ void iln_finalize(const float* sums, const float* ssqs, float* in_m, float* in_is,
                             float* ln_m, float* ln_is, int C, int HW) {
  int b = blockIdx.x, c = threadIdx.x;
  float s = sums[b * C + c], ss = ssqs[b * C + c];
  float n = (float)HW;
  float m = s / n;
  float var = (ss - s * s / n) / (n - 1.f);   // ddof=1
  in_m[b * C + c]  = m;
  in_is[b * C + c] = rsqrtf(var + EPSN);
  __shared__ float shs[128], shq[128];
  shs[c] = s; shq[c] = ss;
  __syncthreads();
  for (int o = C >> 1; o > 0; o >>= 1) {
    if (c < o) { shs[c] += shs[c + o]; shq[c] += shq[c + o]; }
    __syncthreads();
  }
  if (c == 0) {
    float Nt = n * C;
    float M  = shs[0] / Nt;
    float V  = (shq[0] - shs[0] * shs[0] / Nt) / (Nt - 1.f);
    ln_m[b]  = M;
    ln_is[b] = rsqrtf(V + EPSN);
  }
}

// mode 0: SiLU(iln);  mode 1: sigmoid(iln) * gate
__global__ void iln_act_out(const float* __restrict__ raw, const float* in_m, const float* in_is,
                            const float* ln_m, const float* ln_is,
                            const float* rho, const float* gamma, const float* beta,
                            const float* __restrict__ gate, float* __restrict__ out,
                            int C, int HW, int chanOff, int outC, int mode) {
  int i = blockIdx.x * blockDim.x + threadIdx.x;
  int total = 2 * C * HW;
  if (i >= total) return;
  int p = i % HW;
  int c = (i / HW) % C;
  int b = i / (HW * C);
  float x = raw[i];
  float r = rho[c];
  float v = r * (x - in_m[b * C + c]) * in_is[b * C + c]
          + (1.f - r) * (x - ln_m[b]) * ln_is[b];
  v = v * gamma[c] + beta[c];
  float sg = 1.f / (1.f + expf(-v));
  v = (mode == 0) ? v * sg : sg * gate[i];
  out[((size_t)(b * outC + chanOff + c)) * HW + p] = v;
}

// posMajor=1 -> [b,p,c] (for q/k fragments); posMajor=0 -> [b,c,p] (for V tiles)
__global__ void in_silu_pack(const float* __restrict__ raw, const float* mean, const float* invstd,
                             unsigned short* __restrict__ dst, int C, int HW, int posMajor) {
  int i = blockIdx.x * blockDim.x + threadIdx.x;
  int total = 2 * C * HW;
  if (i >= total) return;
  int p = i % HW;
  int c = (i / HW) % C;
  int b = i / (HW * C);
  float x = raw[i];
  float y = (x - mean[b * C + c]) * invstd[b * C + c];
  y = y / (1.f + expf(-y));
  size_t o = posMajor ? ((size_t)(b * HW + p) * C + c) : (size_t)i;
  dst[o] = f2bf_bits(y);
}

// ---------------------------------------------------------------------------
extern "C" void kernel_launch(void* const* d_in, const int* in_sizes, int n_in,
                              void* d_out, int out_size, void* d_ws, size_t ws_size,
                              hipStream_t stream) {
  (void)in_sizes; (void)n_in; (void)out_size; (void)ws_size;
  const float* y     = (const float*)d_in[0];
  const float* s     = (const float*)d_in[1];
  const float* w_bs  = (const float*)d_in[2];
  const float* w_by  = (const float*)d_in[3];
  const float* w_pu  = (const float*)d_in[4];
  const float* rho_p = (const float*)d_in[5];
  const float* gam_p = (const float*)d_in[6];
  const float* bet_p = (const float*)d_in[7];
  const float* w_gr  = (const float*)d_in[8];
  const float* rho_g = (const float*)d_in[9];
  const float* gam_g = (const float*)d_in[10];
  const float* bet_g = (const float*)d_in[11];
  float* out = (float*)d_out;

  char* base = (char*)d_ws;
  size_t off = 0;
  auto alloc = [&](size_t bytes) -> void* {
    void* p = base + off;
    off = (off + bytes + 255) & ~(size_t)255;
    return p;
  };
  float* g_raw = (float*)alloc(2ull * 128 * 16384 * 4);
  float* p_raw = (float*)alloc(2ull * 128 * 16384 * 4);
  float* q_raw = (float*)alloc(2ull * 128 * 4096 * 4);
  float* v_raw = (float*)alloc(2ull * 128 * 4096 * 4);
  float* mha2d = (float*)alloc(2ull * 128 * 4096 * 4);
  unsigned short* q_bf   = (unsigned short*)alloc(2ull * 4096 * 128 * 2);
  unsigned short* vT_bf  = (unsigned short*)alloc(2ull * 128 * 4096 * 2);
  unsigned short* wgr_bf = (unsigned short*)alloc(128ull * 2304 * 2);
  unsigned short* wby_bf = (unsigned short*)alloc(128ull * 2304 * 2);
  unsigned short* wbs_bf = (unsigned short*)alloc(128ull * 1152 * 2);
  unsigned short* wpu_bf = (unsigned short*)alloc(128ull * 1152 * 2);
  float* st_sum = (float*)alloc(256 * 4);
  float* st_ssq = (float*)alloc(256 * 4);
  float* nm_m   = (float*)alloc(256 * 4);
  float* nm_is  = (float*)alloc(256 * 4);
  float* ln_m   = (float*)alloc(64 * 4);
  float* ln_is  = (float*)alloc(64 * 4);

  // weights -> bf16 (OIHW flat == [co][k] with k = ci*9 + kh*3 + kw)
  cvt_bf16<<<(294912 + 255) / 256, 256, 0, stream>>>(w_gr, wgr_bf, 294912);
  cvt_bf16<<<(294912 + 255) / 256, 256, 0, stream>>>(w_by, wby_bf, 294912);
  cvt_bf16<<<(147456 + 255) / 256, 256, 0, stream>>>(w_bs, wbs_bf, 147456);
  cvt_bf16<<<(147456 + 255) / 256, 256, 0, stream>>>(w_pu, wpu_bf, 147456);

  // green branch: conv(upsample2(y)) -> ILN -> SiLU -> out[:,128:256]
  conv3x3_wmma<<<2 * 128 * 2, 128, 0, stream>>>(y, wgr_bf, g_raw, 256, 128, 128, 128, 128, 1, 1, 0);
  sum_stats<<<256, 256, 0, stream>>>(g_raw, st_sum, st_ssq, 16384);
  iln_finalize<<<2, 128, 0, stream>>>(st_sum, st_ssq, nm_m, nm_is, ln_m, ln_is, 128, 16384);
  iln_act_out<<<(2 * 128 * 16384 + 255) / 256, 256, 0, stream>>>(
      g_raw, nm_m, nm_is, ln_m, ln_is, rho_g, gam_g, bet_g, nullptr, out, 128, 16384, 128, 256, 0);

  // blue convs with PE folded into A-tile gather
  conv3x3_wmma<<<2 * 64 * 1, 128, 0, stream>>>(y, wby_bf, q_raw, 256, 64, 64, 64, 64, 1, 0, 1);
  conv3x3_wmma<<<2 * 64 * 1, 128, 0, stream>>>(s, wbs_bf, v_raw, 128, 128, 128, 64, 64, 2, 0, 1);
  in_stats<<<256, 256, 0, stream>>>(q_raw, nm_m, nm_is, 4096);
  in_silu_pack<<<(2 * 128 * 4096 + 255) / 256, 256, 0, stream>>>(q_raw, nm_m, nm_is, q_bf, 128, 4096, 1);
  in_stats<<<256, 256, 0, stream>>>(v_raw, nm_m, nm_is, 4096);
  in_silu_pack<<<(2 * 128 * 4096 + 255) / 256, 256, 0, stream>>>(v_raw, nm_m, nm_is, vT_bf, 128, 4096, 0);

  // MHA(q, q, v), 4 heads, flash-style
  flash_attn<<<2 * 4 * 64, 128, 0, stream>>>(q_bf, vT_bf, mha2d, 4096, 128, 4);

  // purple branch: conv(upsample2(mha)) -> ILN -> sigmoid * s -> out[:,0:128]
  conv3x3_wmma<<<2 * 128 * 2, 128, 0, stream>>>(mha2d, wpu_bf, p_raw, 128, 128, 128, 128, 128, 1, 1, 0);
  sum_stats<<<256, 256, 0, stream>>>(p_raw, st_sum, st_ssq, 16384);
  iln_finalize<<<2, 128, 0, stream>>>(st_sum, st_ssq, nm_m, nm_is, ln_m, ln_is, 128, 16384);
  iln_act_out<<<(2 * 128 * 16384 + 255) / 256, 256, 0, stream>>>(
      p_raw, nm_m, nm_is, ln_m, ln_is, rho_p, gam_p, bet_p, s, out, 128, 16384, 0, 256, 1);
}
